// Mlp_moe_30648886624262
// MI455X (gfx1250) — matrix operand order repaired and checked by reading, not compiled, not run on hardware
//
#include <hip/hip_runtime.h>
#include <hip/hip_bf16.h>
#include <math.h>

typedef __bf16 bf16_t;
typedef __attribute__((ext_vector_type(16))) __bf16 v16bf;
typedef __attribute__((ext_vector_type(8)))  __bf16 v8bf;
typedef __attribute__((ext_vector_type(4)))  __bf16 v4bf;
typedef __attribute__((ext_vector_type(8)))  float  v8f;

#define D_DIM   768
#define H_DIM   3072
#define B_SZ    64
#define NPATCH  197
#define NTOK    203
#define NCLS    6
#define M_PATCH (B_SZ * NPATCH)   // 12608 = 197 * 64

__device__ __forceinline__ float gelu_exact(float v) {
    return 0.5f * v * (1.0f + erff(v * 0.70710678118654752f));
}

// gfx1250 async copy: global -> LDS, 16 bytes per lane, tracked by ASYNCcnt.
__device__ __forceinline__ void async_g2l_b128(unsigned lds_off, const void* gp) {
    asm volatile("global_load_async_to_lds_b128 %0, %1, off"
                 :: "v"(lds_off), "v"((unsigned long long)(uintptr_t)gp)
                 : "memory");
}
__device__ __forceinline__ void wait_async0() {
    asm volatile("s_wait_asynccnt 0x0" ::: "memory");
}

// ---------------------------------------------------------------------------
// Prologue converts: fp32 -> bf16 (weights verbatim, patches gathered so the
// GEMM A matrix is dense row-major [12608][768]).
// ---------------------------------------------------------------------------
__global__ __launch_bounds__(256) void k_cvt_w(const float* __restrict__ src,
                                               bf16_t* __restrict__ dst, int n4) {
    const int i = blockIdx.x * 256 + threadIdx.x;
    if (i >= n4) return;
    const float4 v = ((const float4*)src)[i];
    const v4bf o = {(bf16_t)v.x, (bf16_t)v.y, (bf16_t)v.z, (bf16_t)v.w};
    *(v4bf*)(dst + (size_t)i * 4) = o;
}

__global__ __launch_bounds__(256) void k_cvt_x(const float* __restrict__ x,
                                               bf16_t* __restrict__ a1) {
    const int row4 = D_DIM / 4;                       // 192
    const int i = blockIdx.x * 256 + threadIdx.x;
    if (i >= M_PATCH * row4) return;
    const int m = i / row4, c = i - m * row4;
    const int b = m / NPATCH, p = m - b * NPATCH;
    const float4 v = *(const float4*)(x + (size_t)(b * NTOK + NCLS + p) * D_DIM + c * 4);
    const v4bf o = {(bf16_t)v.x, (bf16_t)v.y, (bf16_t)v.z, (bf16_t)v.w};
    *(v4bf*)(a1 + (size_t)m * D_DIM + c * 4) = o;
}

// ---------------------------------------------------------------------------
// GEMM1: hid = GELU(A[12608,768] x W1^T[768,3072] + b1), bf16 out.
// 4-wave block computes a 64x64 tile; B k-chunks (64 rows x 32 k, bf16, 4KB)
// are async-staged into double-buffered LDS; each wave holds 4 accumulators.
// ---------------------------------------------------------------------------
__global__ __launch_bounds__(128) void k_gemm1(const bf16_t* __restrict__ A,
                                               const bf16_t* __restrict__ W,
                                               const float* __restrict__ bias,
                                               bf16_t* __restrict__ hid) {
    __shared__ __align__(16) bf16_t btile[2][64][32];

    const int tix  = threadIdx.x;
    const int wav  = tix >> 5;
    const int lane = tix & 31;
    const int l15  = lane & 15;
    const int hi   = lane >> 4;

    const int NT = H_DIM / 64;                        // 48 N-tiles
    const int tm = blockIdx.x / NT;
    const int tn = blockIdx.x - tm * NT;
    const int m0 = tm * 64, n0 = tn * 64;

    const int kbA = hi * 8;                           // ISA 16-bit A half-wave K base
    const int kbB = hi * 8;                           // ISA 16-bit B half-wave K base
    const bf16_t* __restrict__ arow = A + (size_t)(m0 + wav * 16 + l15) * D_DIM;

    auto stage = [&](int buf, int k0) {               // 128 thr x 2 x 16B = 4KB chunk
        #pragma unroll
        for (int s = 0; s < 2; ++s) {
            const int c = tix + s * 128;              // 0..255
            const int j = c >> 2;                     // B row (n-local) 0..63
            const int q = c & 3;                      // 16B sub-chunk
            const bf16_t* g = W + (size_t)(n0 + j) * D_DIM + k0 + q * 8;
            async_g2l_b128((unsigned)(uintptr_t)&btile[buf][j][q * 8], g);
        }
    };

    v8f acc[4] = {{}, {}, {}, {}};
    stage(0, 0);
    const int KSTEPS = D_DIM / 32;                    // 24
    for (int it = 0; it < KSTEPS; ++it) {
        const int k0 = it * 32;
        wait_async0();                                // my stage of buf cur done
        __syncthreads();                              // everyone's stage done
        if (it + 1 < KSTEPS) stage((it + 1) & 1, k0 + 32);   // overlaps compute
        const int buf = it & 1;

        if (k0 + 64 < D_DIM) __builtin_prefetch(arow + k0 + 64, 0, 1);
        const v8bf a0 = *(const v8bf*)(arow + k0 + kbA);
        const v8bf a1 = *(const v8bf*)(arow + k0 + kbA + 16);
        const v16bf af = __builtin_shufflevector(
            a0, a1, 0, 1, 2, 3, 4, 5, 6, 7, 8, 9, 10, 11, 12, 13, 14, 15);

        #pragma unroll
        for (int j = 0; j < 4; ++j) {
            const v8bf b0 = *(const v8bf*)&btile[buf][j * 16 + l15][kbB];
            const v8bf b1 = *(const v8bf*)&btile[buf][j * 16 + l15][kbB + 16];
            const v16bf bfrag = __builtin_shufflevector(
                b0, b1, 0, 1, 2, 3, 4, 5, 6, 7, 8, 9, 10, 11, 12, 13, 14, 15);
            acc[j] = __builtin_amdgcn_wmma_f32_16x16x32_bf16(
                         false, af, false, bfrag, (short)0, acc[j], false, false);
        }
    }

    #pragma unroll
    for (int j = 0; j < 4; ++j) {
        const int n = n0 + j * 16 + l15;
        const float bv = bias[n];
        #pragma unroll
        for (int r = 0; r < 8; ++r) {                 // C/D layout: M = r + 8*hi
            const int mr = m0 + wav * 16 + r + hi * 8;
            hid[(size_t)mr * H_DIM + n] = (bf16_t)gelu_exact(acc[j][r] + bv);
        }
    }
}

// ---------------------------------------------------------------------------
// GEMM2: out = hid[12608,3072] x W2^T[3072,768] + b2, fp32 scattered into
// d_out[b, 6+p, :]. Same 64x64 tiling / async double-buffered B staging.
// ---------------------------------------------------------------------------
__global__ __launch_bounds__(128) void k_gemm2(const bf16_t* __restrict__ A,
                                               const bf16_t* __restrict__ W,
                                               const float* __restrict__ bias,
                                               float* __restrict__ out) {
    __shared__ __align__(16) bf16_t btile[2][64][32];

    const int tix  = threadIdx.x;
    const int wav  = tix >> 5;
    const int lane = tix & 31;
    const int l15  = lane & 15;
    const int hi   = lane >> 4;

    const int NT = D_DIM / 64;                        // 12 N-tiles
    const int tm = blockIdx.x / NT;
    const int tn = blockIdx.x - tm * NT;
    const int m0 = tm * 64, n0 = tn * 64;

    const int kbA = hi * 8;
    const int kbB = hi * 8;
    const bf16_t* __restrict__ arow = A + (size_t)(m0 + wav * 16 + l15) * H_DIM;

    auto stage = [&](int buf, int k0) {
        #pragma unroll
        for (int s = 0; s < 2; ++s) {
            const int c = tix + s * 128;
            const int j = c >> 2;
            const int q = c & 3;
            const bf16_t* g = W + (size_t)(n0 + j) * H_DIM + k0 + q * 8;
            async_g2l_b128((unsigned)(uintptr_t)&btile[buf][j][q * 8], g);
        }
    };

    v8f acc[4] = {{}, {}, {}, {}};
    stage(0, 0);
    const int KSTEPS = H_DIM / 32;                    // 96
    for (int it = 0; it < KSTEPS; ++it) {
        const int k0 = it * 32;
        wait_async0();
        __syncthreads();
        if (it + 1 < KSTEPS) stage((it + 1) & 1, k0 + 32);
        const int buf = it & 1;

        if (k0 + 64 < H_DIM) __builtin_prefetch(arow + k0 + 64, 0, 1);
        const v8bf a0 = *(const v8bf*)(arow + k0 + kbA);
        const v8bf a1 = *(const v8bf*)(arow + k0 + kbA + 16);
        const v16bf af = __builtin_shufflevector(
            a0, a1, 0, 1, 2, 3, 4, 5, 6, 7, 8, 9, 10, 11, 12, 13, 14, 15);

        #pragma unroll
        for (int j = 0; j < 4; ++j) {
            const v8bf b0 = *(const v8bf*)&btile[buf][j * 16 + l15][kbB];
            const v8bf b1 = *(const v8bf*)&btile[buf][j * 16 + l15][kbB + 16];
            const v16bf bfrag = __builtin_shufflevector(
                b0, b1, 0, 1, 2, 3, 4, 5, 6, 7, 8, 9, 10, 11, 12, 13, 14, 15);
            acc[j] = __builtin_amdgcn_wmma_f32_16x16x32_bf16(
                         false, af, false, bfrag, (short)0, acc[j], false, false);
        }
    }

    #pragma unroll
    for (int j = 0; j < 4; ++j) {
        const int n = n0 + j * 16 + l15;
        const float bv = bias[n];
        #pragma unroll
        for (int r = 0; r < 8; ++r) {
            const int mr = m0 + wav * 16 + r + hi * 8;
            const int bb = mr / NPATCH;
            const int pp = mr - bb * NPATCH;
            out[(size_t)(bb * NTOK + NCLS + pp) * D_DIM + n] = acc[j][r] + bv;
        }
    }
}

// ---------------------------------------------------------------------------
// Cls/MoE path: one block per (batch, cls) token; only the argmax route's
// expert contributes to the output.
// ---------------------------------------------------------------------------
__device__ __forceinline__ float block_reduce(float v, float* s) {
    const int tid = threadIdx.x;
    s[tid] = v;
    __syncthreads();
    for (int off = 128; off > 0; off >>= 1) {
        if (tid < off) s[tid] += s[tid + off];
        __syncthreads();
    }
    const float r = s[0];
    __syncthreads();
    return r;
}

__global__ __launch_bounds__(256) void k_cls(const float* __restrict__ x,
                                             const float* __restrict__ gate_pair,
                                             const float* __restrict__ atom_in_w,
                                             const float* __restrict__ atom_in_b,
                                             const float* __restrict__ atom_out_w,
                                             const float* __restrict__ atom_out_b,
                                             const float* __restrict__ balance_bias,
                                             float* __restrict__ out) {
    const int bn  = blockIdx.x;
    const int b   = bn / NCLS;
    const int n   = bn - b * NCLS;
    const int tid = threadIdx.x;

    __shared__ float s_cls[D_DIM];
    __shared__ float s_hid[H_DIM];
    __shared__ float s_red[256];

    const int SRCT[6][2] = {{0,3},{0,4},{1,3},{1,4},{2,3},{2,4}};
    const int DSTT[6][2] = {{3,0},{4,0},{3,1},{4,1},{3,2},{4,2}};

    const float* crow = x + (size_t)(b * NTOK + n) * D_DIM;
    for (int i = tid; i < D_DIM; i += 256) s_cls[i] = crow[i];
    __syncthreads();

    float p = 0.f;
    for (int i = tid; i < D_DIM; i += 256) { const float v = s_cls[i]; p += v * v; }
    const float ss_cls = block_reduce(p, s_red);
    const float inv_c  = 1.0f / fmaxf(sqrtf(ss_cls), 1e-12f);

    float logit[2];
    #pragma unroll
    for (int r = 0; r < 2; ++r) {
        const float* gw = gate_pair + ((size_t)n * 2 + r) * D_DIM;
        float pd = 0.f, pq = 0.f;
        for (int i = tid; i < D_DIM; i += 256) {
            const float w = gw[i];
            pd += w * s_cls[i];
            pq += w * w;
        }
        const float dot = block_reduce(pd, s_red);
        const float ssw = block_reduce(pq, s_red);
        logit[r] = dot * inv_c * (1.0f / fmaxf(sqrtf(ssw), 1e-12f))
                 + balance_bias[n * 2 + r];
    }
    const int chosen = (logit[1] > logit[0]) ? 1 : 0;      // first index on tie
    const float mx = fmaxf(logit[0], logit[1]);
    const float e0 = __expf(logit[0] - mx);
    const float e1 = __expf(logit[1] - mx);
    const float cw = (chosen ? e1 : e0) / (e0 + e1);

    const int a_src = SRCT[n][chosen];
    const int a_dst = DSTT[n][chosen];

    const float* win = atom_in_w + (size_t)a_src * H_DIM * D_DIM;
    for (int h = tid; h < H_DIM; h += 256) {
        const float* wr = win + (size_t)h * D_DIM;
        float a = 0.f;
        for (int k = 0; k < D_DIM; ++k) a += wr[k] * s_cls[k];
        s_hid[h] = gelu_exact(a + atom_in_b[a_src * H_DIM + h]);
    }
    __syncthreads();

    const float* wout = atom_out_w + (size_t)a_dst * D_DIM * H_DIM;
    float* orow = out + (size_t)(b * NTOK + n) * D_DIM;
    for (int d = tid; d < D_DIM; d += 256) {
        const float* wr = wout + (size_t)d * H_DIM;
        float a = 0.f;
        for (int k = 0; k < H_DIM; ++k) a += wr[k] * s_hid[k];
        orow[d] = (a + atom_out_b[a_dst * D_DIM + d]) * cw;
    }
}

// ---------------------------------------------------------------------------
extern "C" void kernel_launch(void* const* d_in, const int* in_sizes, int n_in,
                              void* d_out, int out_size, void* d_ws, size_t ws_size,
                              hipStream_t stream) {
    const float* x            = (const float*)d_in[0];
    const float* fc1_w        = (const float*)d_in[1];
    const float* fc1_b        = (const float*)d_in[2];
    const float* fc2_w        = (const float*)d_in[3];
    const float* fc2_b        = (const float*)d_in[4];
    const float* gate_pair    = (const float*)d_in[5];
    const float* atom_in_w    = (const float*)d_in[6];
    const float* atom_in_b    = (const float*)d_in[7];
    const float* atom_out_w   = (const float*)d_in[8];
    const float* atom_out_b   = (const float*)d_in[9];
    const float* balance_bias = (const float*)d_in[10];
    float* out = (float*)d_out;

    // Workspace layout (bf16): A1 | W1 | W2 | HID  (~106 MB total)
    bf16_t* A1  = (bf16_t*)d_ws;
    bf16_t* W1  = A1 + (size_t)M_PATCH * D_DIM;
    bf16_t* W2  = W1 + (size_t)H_DIM * D_DIM;
    bf16_t* HID = W2 + (size_t)D_DIM * H_DIM;

    (void)in_sizes; (void)n_in; (void)out_size; (void)ws_size;

    // fp32 -> bf16 prologue
    {
        const int n4x = M_PATCH * (D_DIM / 4);               // 2,420,736
        k_cvt_x<<<(n4x + 255) / 256, 256, 0, stream>>>(x, A1);
        const int n4w = (H_DIM * D_DIM) / 4;                 // 589,824
        k_cvt_w<<<(n4w + 255) / 256, 256, 0, stream>>>(fc1_w, W1, n4w);
        k_cvt_w<<<(n4w + 255) / 256, 256, 0, stream>>>(fc2_w, W2, n4w);
    }
    // GEMM1 + GELU: 197 x 48 blocks of 64x64
    k_gemm1<<<197 * 48, 128, 0, stream>>>(A1, W1, fc1_b, HID);
    // GEMM2 + bias: 197 x 12 blocks of 64x64
    k_gemm2<<<197 * 12, 128, 0, stream>>>(HID, W2, fc2_b, out);
    // Cls/MoE path
    k_cls<<<B_SZ * NCLS, 256, 0, stream>>>(x, gate_pair, atom_in_w, atom_in_b,
                                           atom_out_w, atom_out_b, balance_bias,
                                           out);
}
